// Attention_47064251630340
// MI455X (gfx1250) — matrix operand order repaired and checked by reading, not compiled, not run on hardware
//
#include <hip/hip_runtime.h>

typedef _Float16 h16;
typedef __attribute__((ext_vector_type(8)))  _Float16 v8h;
typedef __attribute__((ext_vector_type(16))) _Float16 v16h;
typedef __attribute__((ext_vector_type(8)))  float    v8f;

#define DIMC   576
#define HN     18
#define KD     32
#define DVDIM  128
#define QKV_N  3456     // H*(2*KD+DV) = 18*192
#define HSTR   192
#define NSEQ   196
#define BATCH  128
#define DH     2304     // H*DV
#define NPAD   224      // 14 tiles of 16, mult of 32
#define MROWS  (BATCH*NSEQ)  // 25088
#define SCALE_Q 0.17677669529663687f   // 32^-0.5

static __device__ __forceinline__ v16h combine8(v8h lo, v8h hi) {
  return __builtin_shufflevector(lo, hi, 0,1,2,3,4,5,6,7,8,9,10,11,12,13,14,15);
}
static __device__ __forceinline__ v8f wmma16(v16h a, v16h b, v8f c) {
  return __builtin_amdgcn_wmma_f32_16x16x32_f16(false, a, false, b, (short)0, c, false, false);
}
// scheduling fence: stops the scheduler from hoisting later memory ops above
// this point (caps load-prefetch depth / register pressure); emits nothing.
static __device__ __forceinline__ void sched_fence() {
  asm volatile("" ::: "memory");
}

// ---------------- f32 -> f16 convert ----------------
__global__ void cvt_f2h(const float* __restrict__ src, h16* __restrict__ dst, int n) {
  int stride = gridDim.x * blockDim.x;
  for (int i = blockIdx.x * blockDim.x + threadIdx.x; i < n; i += stride)
    dst[i] = (h16)src[i];
}

// ---------------- expand bias: Bt[h][m][n] = biases[h, bidx[n, m]] ----------------
// Transposed (n contiguous) so a WMMA C-tile lane reads its 8 row-bias values
// as one 32B vector load that seeds the WMMA C operand directly.
__global__ void expand_bias(const float* __restrict__ biases, const int* __restrict__ bidx,
                            float* __restrict__ bt, int n_off) {
  int i = blockIdx.x * blockDim.x + threadIdx.x;
  if (i >= HN * NSEQ * NPAD) return;
  int n = i % NPAD;
  int m = (i / NPAD) % NSEQ;
  int h = i / (NPAD * NSEQ);
  int nc = n < NSEQ ? n : NSEQ - 1;
  bt[i] = biases[h * n_off + bidx[nc * NSEQ + m]];
}

// ---------------- LayerNorm (one wave per row of 576) ----------------
__global__ __launch_bounds__(128) void ln_kernel(const float* __restrict__ x,
                                                 const float* __restrict__ g,
                                                 const float* __restrict__ b,
                                                 h16* __restrict__ xn) {
  int wid  = blockIdx.x * 4 + (threadIdx.x >> 5);
  int lane = threadIdx.x & 31;
  const float* xr = x + (size_t)wid * DIMC;
  float v[18];
  float s = 0.f;
  #pragma unroll
  for (int i = 0; i < 18; i++) { v[i] = xr[lane + i * 32]; s += v[i]; }
  #pragma unroll
  for (int d = 16; d >= 1; d >>= 1) s += __shfl_xor(s, d, 32);
  float mu = s * (1.f / 576.f);
  float vs = 0.f;
  #pragma unroll
  for (int i = 0; i < 18; i++) { float dd = v[i] - mu; vs += dd * dd; }
  #pragma unroll
  for (int d = 16; d >= 1; d >>= 1) vs += __shfl_xor(vs, d, 32);
  float rs = rsqrtf(vs * (1.f / 576.f) + 1e-5f);
  h16* o = xn + (size_t)wid * DIMC;
  #pragma unroll
  for (int i = 0; i < 18; i++) {
    int idx = lane + i * 32;
    o[idx] = (h16)((v[i] - mu) * rs * g[idx] + b[idx]);
  }
}

// ---------------- WMMA GEMM: C[M,N] = (A[M,K] * W[N,K]^T + bias) * colmul ----
// Block = 8 waves sharing one B-strip (16*NSTRIP cols) over 8 consecutive
// 16-row M-tiles: B fragments hit the WGP$ 8x, cutting L2 traffic ~5x.
// QSCALE: multiply q-columns (col%192 < 32) by 32^-0.5 so attention needs no
// post-scale (its WMMA C seed is the raw bias).
template<bool HALF_OUT, int NSTRIP, bool QSCALE>
__global__ __launch_bounds__(256) void gemm_wmma(const h16* __restrict__ A,
                                                 const h16* __restrict__ W,
                                                 const float* __restrict__ bias,
                                                 void* __restrict__ outp,
                                                 int M, int K, int N) {
  int lane = threadIdx.x & 31;
  int wave = threadIdx.x >> 5;
  int nStrips = N / (16 * NSTRIP);
  int ns = blockIdx.x % nStrips;
  int mb = blockIdx.x / nStrips;
  int mTile = mb * 8 + wave;
  if (mTile * 16 >= M) return;
  int r = lane & 15, hw = lane >> 4;
  int kc8 = hw * 8, kb16 = hw * 16, rbase = hw * 8;
  const h16* Arow = A + (size_t)(mTile * 16 + r) * K;
  v8f acc[NSTRIP];
  #pragma unroll
  for (int t = 0; t < NSTRIP; t++) acc[t] = v8f{0.f};
  #pragma unroll 2
  for (int k0 = 0; k0 < K; k0 += 32) {
    v8h alo = *(const v8h*)(Arow + k0 + kb16);
    v8h ahi = *(const v8h*)(Arow + k0 + kb16 + 8);
    v16h a = combine8(alo, ahi);
    #pragma unroll
    for (int t = 0; t < NSTRIP; t++) {
      int ncol = (ns * NSTRIP + t) * 16 + r;
      v16h bf = *(const v16h*)(W + (size_t)ncol * K + k0 + kb16);
      acc[t] = wmma16(a, bf, acc[t]);
    }
  }
  #pragma unroll
  for (int t = 0; t < NSTRIP; t++) {
    int col = (ns * NSTRIP + t) * 16 + r;
    float bv = bias ? bias[col] : 0.f;
    float mult = 1.f;
    if (QSCALE) mult = ((col % HSTR) < KD) ? SCALE_Q : 1.f;
    #pragma unroll
    for (int j = 0; j < 8; j++) {
      size_t row = (size_t)(mTile * 16 + rbase + j);
      float v = (acc[t][j] + bv) * mult;
      if (HALF_OUT) ((h16*)outp)[row * N + col] = (h16)v;
      else          ((float*)outp)[row * N + col] = v;
    }
  }
}

// ---------------- Attention: one block per (b,h), 4 waves ----------------
__global__ __launch_bounds__(128) void attn_kernel(const h16* __restrict__ qkv,
                                                   const float* __restrict__ bt,
                                                   h16* __restrict__ outp) {
  __shared__ h16 Vt[DVDIM * NPAD];        // V transposed: [d][m], 57344 B
  __shared__ h16 Psm[4 * 16 * NPAD];      // per-wave softmax(P) tiles, 28672 B

  int bh = blockIdx.x;
  int b  = bh / HN;
  int h  = bh - b * HN;
  int tid = threadIdx.x, wave = tid >> 5, lane = tid & 31;
  const h16* base = qkv + (size_t)b * NSEQ * QKV_N + (size_t)h * HSTR;
  const float* bhead = bt + (size_t)h * NSEQ * NPAD;   // [m][n], n contiguous

  // stage V transposed into LDS (zero-pad m >= 196)
  for (int i = tid; i < DVDIM * NPAD; i += 128) {
    int m = i >> 7, d = i & 127;
    h16 val = (h16)0.f;
    if (m < NSEQ) val = base[(size_t)m * QKV_N + 2 * KD + d];
    Vt[d * NPAD + m] = val;
  }
  __syncthreads();

  int r = lane & 15, hw = lane >> 4;
  int kc8 = hw * 8, kb16 = hw * 16, rbase = hw * 8;
  h16* P = Psm + wave * 16 * NPAD;

  for (int nt = wave; nt < 13; nt += 4) {
    int n0 = nt * 16;
    // Q fragment (K = 32, single step); q is pre-scaled by 32^-0.5
    int nrow = n0 + r; if (nrow > NSEQ - 1) nrow = NSEQ - 1;
    const h16* pq = base + (size_t)nrow * QKV_N;
    v16h qa = combine8(*(const v8h*)(pq + kb16), *(const v8h*)(pq + kb16 + 8));

    // scores: tiles 0..11 fully in range -> WMMA result IS s[mt] (no epilogue);
    // tile 12 straddles N=196; tile 13 is entirely out of range (no WMMA).
    // Fully unrolled (constant s[] indices -> registers); sched_fence every
    // 4 tiles caps in-flight loads so allocation stays under 256 VGPRs.
    v8f s[13];
    #pragma unroll
    for (int mt = 0; mt < 12; mt++) {
      int mc = mt * 16 + r;
      v16h kf = *(const v16h*)(base + (size_t)mc * QKV_N + KD + kb16);
      v8f c = *(const v8f*)(bhead + (size_t)mc * NPAD + n0 + rbase);
      s[mt] = wmma16(qa, kf, c);
      if ((mt & 3) == 3) sched_fence();
    }
    {
      int mcol = 192 + r;
      int mc = mcol > NSEQ - 1 ? NSEQ - 1 : mcol;
      v16h kf = *(const v16h*)(base + (size_t)mc * QKV_N + KD + kb16);
      v8f c = *(const v8f*)(bhead + (size_t)mc * NPAD + n0 + rbase);
      c = wmma16(qa, kf, c);
      #pragma unroll
      for (int j = 0; j < 8; j++) s[12][j] = (mcol > NSEQ - 1) ? -1e30f : c[j];
    }

    // softmax over 13 tiles (C layout: row j lives in one lane-half)
    #pragma unroll
    for (int j = 0; j < 8; j++) {
      float mx = s[0][j];
      #pragma unroll
      for (int t = 1; t < 13; t++) mx = fmaxf(mx, s[t][j]);
      #pragma unroll
      for (int d = 1; d < 16; d <<= 1) mx = fmaxf(mx, __shfl_xor(mx, d, 16));
      float su = 0.f;
      #pragma unroll
      for (int t = 0; t < 13; t++) { float e = __expf(s[t][j] - mx); s[t][j] = e; su += e; }
      #pragma unroll
      for (int d = 1; d < 16; d <<= 1) su += __shfl_xor(su, d, 16);
      float inv = 1.f / su;
      #pragma unroll
      for (int t = 0; t < 13; t++)
        P[(rbase + j) * NPAD + t * 16 + r] = (h16)(s[t][j] * inv);
      P[(rbase + j) * NPAD + 13 * 16 + r] = (h16)0.f;   // padded columns
    }
    asm volatile("s_wait_dscnt 0" ::: "memory");  // per-wave LDS RAW ordering

    // O = P @ V : K dim = 224 in 7 steps of 32, 8 d-tiles
    v8f o[8];
    #pragma unroll
    for (int dt = 0; dt < 8; dt++) o[dt] = v8f{0.f};
    #pragma unroll
    for (int kk = 0; kk < 7; kk++) {
      const h16* pp = P + r * NPAD + kk * 32 + kb16;
      v16h pa = combine8(*(const v8h*)pp, *(const v8h*)(pp + 8));
      #pragma unroll
      for (int dt = 0; dt < 8; dt++) {
        v16h vb = *(const v16h*)(Vt + (dt * 16 + r) * NPAD + kk * 32 + kb16);
        o[dt] = wmma16(pa, vb, o[dt]);
      }
      sched_fence();
    }
    // store O (f16) into attn-out (25088 x 2304)
    #pragma unroll
    for (int dt = 0; dt < 8; dt++) {
      int col = h * DVDIM + dt * 16 + r;
      #pragma unroll
      for (int j = 0; j < 8; j++) {
        int rw = n0 + rbase + j;
        if (rw < NSEQ)
          outp[(size_t)(b * NSEQ + rw) * DH + col] = (h16)o[dt][j];
      }
    }
  }
}

extern "C" void kernel_launch(void* const* d_in, const int* in_sizes, int n_in,
                              void* d_out, int out_size, void* d_ws, size_t ws_size,
                              hipStream_t stream) {
  const float* x          = (const float*)d_in[0];
  const float* ln_g       = (const float*)d_in[1];
  const float* ln_b       = (const float*)d_in[2];
  const float* qkv_w      = (const float*)d_in[3];
  const float* qkv_b      = (const float*)d_in[4];
  const float* proj_w     = (const float*)d_in[5];
  const float* proj_b     = (const float*)d_in[6];
  const float* attn_bias  = (const float*)d_in[7];
  const int*   bias_idxs  = (const int*)d_in[8];
  int n_off = in_sizes[7] / HN;

  char* ws = (char*)d_ws;
  h16* xn   = (h16*)ws; ws += (size_t)MROWS * DIMC * 2;
  h16* qw   = (h16*)ws; ws += (size_t)QKV_N * DIMC * 2;
  h16* pw   = (h16*)ws; ws += (size_t)DIMC * DH * 2;
  h16* qkvb = (h16*)ws; ws += (size_t)MROWS * QKV_N * 2;
  h16* ao   = (h16*)ws; ws += (size_t)MROWS * DH * 2;
  float* bt = (float*)ws; ws += (size_t)HN * NSEQ * NPAD * 4;

  cvt_f2h<<<2048, 256, 0, stream>>>(qkv_w, qw, QKV_N * DIMC);
  cvt_f2h<<<2048, 256, 0, stream>>>(proj_w, pw, DIMC * DH);
  expand_bias<<<(HN * NSEQ * NPAD + 255) / 256, 256, 0, stream>>>(
      attn_bias, bias_idxs, bt, n_off);
  ln_kernel<<<MROWS / 4, 128, 0, stream>>>(x, ln_g, ln_b, xn);

  // QKV GEMM: 25088 x 576 -> 25088 x 3456 (f16 out, q pre-scaled)
  gemm_wmma<true, 8, true><<<(MROWS / 128) * (QKV_N / 128), 256, 0, stream>>>(
      xn, qw, qkv_b, (void*)qkvb, MROWS, DIMC, QKV_N);

  // attention per (b,h)
  attn_kernel<<<BATCH * HN, 128, 0, stream>>>(qkvb, bt, ao);

  // proj GEMM: 25088 x 2304 -> 25088 x 576 (f32 out)
  gemm_wmma<false, 4, false><<<(MROWS / 128) * (DIMC / 64), 256, 0, stream>>>(
      ao, pw, proj_b, d_out, MROWS, DH, DIMC);
}
